// SparseMoE_20761871909664
// MI455X (gfx1250) — matrix operand order repaired and checked by reading, not compiled, not run on hardware
//
#include <hip/hip_runtime.h>
#include <hip/hip_bf16.h>
#include <cstdint>

#define N_TOK 2048
#define DDIM  1024
#define EEXP  8
#define FDIM  4096
#define MT    16
#define MT2   32

typedef __attribute__((ext_vector_type(16))) __bf16 v16bf;
typedef __attribute__((ext_vector_type(8)))  float  v8f;

union FragBF { v16bf v; uint32_t u[8]; uint4 q[2]; };

__device__ __forceinline__ uint32_t bf16_rne(float f) {
    uint32_t u = __float_as_uint(f);
    return (u + 0x7FFFu + ((u >> 16) & 1u)) >> 16;
}
__device__ __forceinline__ uint32_t pk_bf16(float lo, float hi) {
    return bf16_rne(lo) | (bf16_rne(hi) << 16);
}
__device__ __forceinline__ float gelu_erf(float x) {
    return 0.5f * x * (1.0f + erff(x * 0.70710678118654752f));
}

// ---------------------------------------------------------------------------
// Kernel 0: zero output + workspace header (cnt[8] ints, imp[8], load[8])
// ---------------------------------------------------------------------------
__global__ void zero_kernel(float* __restrict__ out, int n, float* __restrict__ wshdr) {
    int i = blockIdx.x * blockDim.x + threadIdx.x;
    for (; i < n; i += gridDim.x * blockDim.x) out[i] = 0.0f;
    if (blockIdx.x == 0 && threadIdx.x < 24) wshdr[threadIdx.x] = 0.0f; // 0.0f bits == 0 for ints
}

// ---------------------------------------------------------------------------
// Kernel 1: gating — one wave per token. logits -> softmax -> top2 -> route.
// ---------------------------------------------------------------------------
__global__ __launch_bounds__(256)
void gate_kernel(const float* __restrict__ x, const float* __restrict__ gw,
                 int* __restrict__ cnt, float* __restrict__ imp, float* __restrict__ loadv,
                 float* __restrict__ combine, int* __restrict__ lists) {
    const int wave = threadIdx.x >> 5;
    const int lane = threadIdx.x & 31;
    const int n = blockIdx.x * 8 + wave;

    float acc[EEXP];
#pragma unroll
    for (int e = 0; e < EEXP; ++e) acc[e] = 0.0f;

    const float* xr = x + (size_t)n * DDIM;
    for (int d = lane; d < DDIM; d += 32) {
        float xv = xr[d];
        const float* g = gw + d * EEXP;
#pragma unroll
        for (int e = 0; e < EEXP; ++e) acc[e] += xv * g[e];
    }
#pragma unroll
    for (int off = 16; off >= 1; off >>= 1) {
#pragma unroll
        for (int e = 0; e < EEXP; ++e) acc[e] += __shfl_xor(acc[e], off, 32);
    }

    if (lane == 0) {
        float mx = acc[0];
#pragma unroll
        for (int e = 1; e < EEXP; ++e) mx = fmaxf(mx, acc[e]);
        float p[EEXP], s = 0.0f;
#pragma unroll
        for (int e = 0; e < EEXP; ++e) { p[e] = expf(acc[e] - mx); s += p[e]; }
        float inv_s = 1.0f / s;
#pragma unroll
        for (int e = 0; e < EEXP; ++e) p[e] *= inv_s;

        int i1 = 0;
#pragma unroll
        for (int e = 1; e < EEXP; ++e) if (p[e] > p[i1]) i1 = e;
        int i2 = (i1 == 0) ? 1 : 0;
#pragma unroll
        for (int e = 0; e < EEXP; ++e) if (e != i1 && p[e] > p[i2]) i2 = e;

        float inv = 1.0f / (p[i1] + p[i2] + 1e-9f);
        float w1 = p[i1] * inv, w2 = p[i2] * inv;
#pragma unroll
        for (int e = 0; e < EEXP; ++e)
            combine[n * EEXP + e] = (e == i1) ? w1 : ((e == i2) ? w2 : 0.0f);

        int pos1 = atomicAdd(&cnt[i1], 1); lists[i1 * N_TOK + pos1] = n;
        int pos2 = atomicAdd(&cnt[i2], 1); lists[i2 * N_TOK + pos2] = n;
#pragma unroll
        for (int e = 0; e < EEXP; ++e) {
            unsafeAtomicAdd(&imp[e], p[e]);
            unsafeAtomicAdd(&loadv[e], (p[e] > 0.0f) ? 1.0f : 0.0f);
        }
    }
}

// ---------------------------------------------------------------------------
// Weight pre-pack: fp32 -> bf16, swizzled into the exact B-fragment register
// image (ISA 7.12.2): frag[lane][v] dword = bf16 pair (k = ks*32+2(v&3)+16(v>>2)+8*hi,
// k+1) at column tile*16 + (lane&15).  One 1KB fragment per (tile, k-step).
// ---------------------------------------------------------------------------
__global__ __launch_bounds__(256)
void pack_w1_kernel(const float* __restrict__ W1, uint32_t* __restrict__ W1p) {
    const size_t total = (size_t)EEXP * (FDIM / 16) * (DDIM / 32) * 256;
    size_t idx = (size_t)blockIdx.x * blockDim.x + threadIdx.x;
    const size_t stride = (size_t)gridDim.x * blockDim.x;
    for (; idx < total; idx += stride) {
        int v    = (int)(idx & 7);
        int lane = (int)((idx >> 3) & 31);
        size_t r = idx >> 8;
        int ks = (int)(r & (DDIM / 32 - 1));   // 32 k-steps
        r >>= 5;
        int ft = (int)(r & (FDIM / 16 - 1));   // 256 f-tiles
        int e  = (int)(r >> 8);
        int k = ks * 32 + 2 * (v & 3) + 16 * (v >> 2) + 8 * (lane >> 4);
        int f = ft * 16 + (lane & 15);
        const float* src = W1 + ((size_t)e * DDIM + k) * FDIM + f;
        W1p[idx] = pk_bf16(src[0], src[FDIM]);
    }
}

__global__ __launch_bounds__(256)
void pack_w2_kernel(const float* __restrict__ W2, uint32_t* __restrict__ W2p) {
    const size_t total = (size_t)EEXP * (DDIM / 16) * (FDIM / 32) * 256;
    size_t idx = (size_t)blockIdx.x * blockDim.x + threadIdx.x;
    const size_t stride = (size_t)gridDim.x * blockDim.x;
    for (; idx < total; idx += stride) {
        int v    = (int)(idx & 7);
        int lane = (int)((idx >> 3) & 31);
        size_t r = idx >> 8;
        int ks = (int)(r & (FDIM / 32 - 1));   // 128 k-steps
        r >>= 7;
        int dt = (int)(r & (DDIM / 16 - 1));   // 64 d-tiles
        int e  = (int)(r >> 6);
        int k = ks * 32 + 2 * (v & 3) + 16 * (v >> 2) + 8 * (lane >> 4);
        int d = dt * 16 + (lane & 15);
        const float* src = W2 + ((size_t)e * FDIM + k) * DDIM + d;
        W2p[idx] = pk_bf16(src[0], src[DDIM]);
    }
}

// ---------------------------------------------------------------------------
// Kernel 2 (fast path): fused expert MLP, M = 32 routed tokens per block.
// Each B fragment (1KB, coalesced global_load_b128 x2) feeds TWO wmma (row
// tiles 0-15 and 16-31), halving weight-stream traffic.  Uses the full
// 320KB WGP LDS: sA 64KB + sH 256KB.
// ---------------------------------------------------------------------------
__global__ __launch_bounds__(256)
void moe_packed_kernel(const float* __restrict__ x,
                       const uint32_t* __restrict__ W1p, const float* __restrict__ b1,
                       const uint32_t* __restrict__ W2p, const float* __restrict__ b2,
                       const int* __restrict__ cnt, const float* __restrict__ combine,
                       const int* __restrict__ lists, float* __restrict__ out) {
    __shared__ uint32_t sA[MT2 * (DDIM / 2)];   // 32 x 1024 bf16 =  64 KB
    __shared__ uint32_t sH[MT2 * (FDIM / 2)];   // 32 x 4096 bf16 = 256 KB

    const int e      = blockIdx.y;
    const int mstart = blockIdx.x * MT2;
    const int ce     = cnt[e];
    if (mstart >= ce) return;

    const int tid  = threadIdx.x;
    const int wave = tid >> 5;
    const int lane = tid & 31;
    const int nL   = lane & 15;
    const int hiL  = lane >> 4;
    const int* mylist = lists + e * N_TOK;

    // gather + convert 32-token x tile to bf16 in LDS
    for (int p = tid; p < MT2 * (DDIM / 2); p += 256) {
        int r   = p >> 9;
        int kp  = p & 511;
        int row = mstart + r;
        int tok = (row < ce) ? mylist[row] : 0;
        const float* xr = x + (size_t)tok * DDIM + 2 * kp;
        sA[p] = pk_bf16(xr[0], xr[1]);
    }
    __syncthreads();

    // ---- GEMM1: 32 f-tiles per wave, 2 row-tiles each ----
    unsigned short* sHs = (unsigned short*)sH;
    for (int i = 0; i < (FDIM / 16) / 8; ++i) {
        const int ft = wave + 8 * i;
        const int f0 = ft * 16;
        const uint32_t* wbase = W1p + (((size_t)e * (FDIM / 16) + ft) * (DDIM / 32)) * 256
                                    + (size_t)lane * 8;
        v8f c0 = {}, c1 = {};
#pragma unroll 2
        for (int kb = 0; kb < DDIM; kb += 32) {
            FragBF a0, a1, b;
            const uint32_t* ar0 = &sA[nL * (DDIM / 2) + (kb >> 1) + 4 * hiL];
            const uint32_t* ar1 = ar0 + 16 * (DDIM / 2);
            a0.q[0] = *(const uint4*)(ar0);
            a0.q[1] = *(const uint4*)(ar0 + 8);
            a1.q[0] = *(const uint4*)(ar1);
            a1.q[1] = *(const uint4*)(ar1 + 8);
            const uint4* bp = (const uint4*)(wbase + (size_t)(kb >> 5) * 256);
            b.q[0] = bp[0];
            b.q[1] = bp[1];
            c0 = __builtin_amdgcn_wmma_f32_16x16x32_bf16(false, a0.v, false, b.v,
                                                         (short)0, c0, false, false);
            c1 = __builtin_amdgcn_wmma_f32_16x16x32_bf16(false, a1.v, false, b.v,
                                                         (short)0, c1, false, false);
        }
        const float bv = b1[e * FDIM + f0 + nL];
#pragma unroll
        for (int v = 0; v < 8; ++v) {
            int m = v + 8 * hiL;
            int f = f0 + nL;
            sHs[m * FDIM + f]        = (unsigned short)bf16_rne(gelu_erf(c0[v] + bv));
            sHs[(16 + m) * FDIM + f] = (unsigned short)bf16_rne(gelu_erf(c1[v] + bv));
        }
    }
    __syncthreads();

    // ---- GEMM2: 8 d-tiles per wave, 2 row-tiles each ----
    for (int j = 0; j < (DDIM / 16) / 8; ++j) {
        const int dt = wave + 8 * j;
        const int d0 = dt * 16;
        const uint32_t* wbase = W2p + (((size_t)e * (DDIM / 16) + dt) * (FDIM / 32)) * 256
                                    + (size_t)lane * 8;
        v8f c0 = {}, c1 = {};
#pragma unroll 2
        for (int kb = 0; kb < FDIM; kb += 32) {
            FragBF a0, a1, b;
            const uint32_t* ar0 = &sH[nL * (FDIM / 2) + (kb >> 1) + 4 * hiL];
            const uint32_t* ar1 = ar0 + 16 * (FDIM / 2);
            a0.q[0] = *(const uint4*)(ar0);
            a0.q[1] = *(const uint4*)(ar0 + 8);
            a1.q[0] = *(const uint4*)(ar1);
            a1.q[1] = *(const uint4*)(ar1 + 8);
            const uint4* bp = (const uint4*)(wbase + (size_t)(kb >> 5) * 256);
            b.q[0] = bp[0];
            b.q[1] = bp[1];
            c0 = __builtin_amdgcn_wmma_f32_16x16x32_bf16(false, a0.v, false, b.v,
                                                         (short)0, c0, false, false);
            c1 = __builtin_amdgcn_wmma_f32_16x16x32_bf16(false, a1.v, false, b.v,
                                                         (short)0, c1, false, false);
        }
        const float b2v = b2[e * DDIM + d0 + nL];
#pragma unroll
        for (int rt = 0; rt < 2; ++rt) {
#pragma unroll
            for (int v = 0; v < 8; ++v) {
                int m   = rt * 16 + v + 8 * hiL;
                int row = mstart + m;
                bool ok = row < ce;
                int tok = ok ? mylist[row] : 0;
                float w = ok ? combine[tok * EEXP + e] : 0.0f;
                float cv = rt ? c1[v] : c0[v];
                float val = (cv + b2v) * w;
                unsafeAtomicAdd(out + (size_t)tok * DDIM + d0 + nL, val);
            }
        }
    }
}

// ---------------------------------------------------------------------------
// Kernel 2 (fallback, small ws): on-the-fly fp32->bf16 conversion, M=16.
// ---------------------------------------------------------------------------
__global__ __launch_bounds__(256)
void moe_kernel(const float* __restrict__ x,  const float* __restrict__ W1,
                const float* __restrict__ b1, const float* __restrict__ W2,
                const float* __restrict__ b2, const int* __restrict__ cnt,
                const float* __restrict__ combine, const int* __restrict__ lists,
                float* __restrict__ out) {
    __shared__ uint32_t sA[MT * (DDIM / 2)];
    __shared__ uint32_t sH[MT * (FDIM / 2)];

    const int e      = blockIdx.y;
    const int mstart = blockIdx.x * MT;
    const int ce     = cnt[e];
    if (mstart >= ce) return;

    const int tid  = threadIdx.x;
    const int wave = tid >> 5;
    const int lane = tid & 31;
    const int nL   = lane & 15;
    const int hiL  = lane >> 4;
    const int* mylist = lists + e * N_TOK;

    for (int p = tid; p < MT * (DDIM / 2); p += 256) {
        int r   = p >> 9;
        int kp  = p & 511;
        int row = mstart + r;
        int tok = (row < ce) ? mylist[row] : 0;
        const float* xr = x + (size_t)tok * DDIM + 2 * kp;
        sA[p] = pk_bf16(xr[0], xr[1]);
    }
    __syncthreads();

    const float* W1e = W1 + (size_t)e * DDIM * FDIM;
    unsigned short* sHs = (unsigned short*)sH;
    for (int i = 0; i < (FDIM / 16) / 8; ++i) {
        const int f0 = (wave + 8 * i) * 16;
        v8f c = {};
        for (int kb = 0; kb < DDIM; kb += 32) {
            FragBF a, b;
            const uint32_t* ar = &sA[nL * (DDIM / 2) + (kb >> 1) + 4 * hiL];
            a.q[0] = *(const uint4*)(ar);
            a.q[1] = *(const uint4*)(ar + 8);
            const float* wp = W1e + (size_t)(kb + 8 * hiL) * FDIM + f0 + nL;
#pragma unroll
            for (int v = 0; v < 8; ++v) {
                const float* q = wp + (size_t)(2 * (v & 3) + 16 * (v >> 2)) * FDIM;
                b.u[v] = pk_bf16(q[0], q[FDIM]);
            }
            c = __builtin_amdgcn_wmma_f32_16x16x32_bf16(false, a.v, false, b.v,
                                                        (short)0, c, false, false);
        }
#pragma unroll
        for (int v = 0; v < 8; ++v) {
            int m = v + 8 * hiL;
            int f = f0 + nL;
            float val = c[v] + b1[e * FDIM + f];
            sHs[m * FDIM + f] = (unsigned short)bf16_rne(gelu_erf(val));
        }
    }
    __syncthreads();

    const float* W2e = W2 + (size_t)e * FDIM * DDIM;
    for (int j = 0; j < (DDIM / 16) / 8; ++j) {
        const int d0 = (wave + 8 * j) * 16;
        v8f c = {};
        for (int kb = 0; kb < FDIM; kb += 32) {
            FragBF a, b;
            const uint32_t* ar = &sH[nL * (FDIM / 2) + (kb >> 1) + 4 * hiL];
            a.q[0] = *(const uint4*)(ar);
            a.q[1] = *(const uint4*)(ar + 8);
            const float* wp = W2e + (size_t)(kb + 8 * hiL) * DDIM + d0 + nL;
#pragma unroll
            for (int v = 0; v < 8; ++v) {
                const float* q = wp + (size_t)(2 * (v & 3) + 16 * (v >> 2)) * DDIM;
                b.u[v] = pk_bf16(q[0], q[DDIM]);
            }
            c = __builtin_amdgcn_wmma_f32_16x16x32_bf16(false, a.v, false, b.v,
                                                        (short)0, c, false, false);
        }
#pragma unroll
        for (int v = 0; v < 8; ++v) {
            int m   = v + 8 * hiL;
            int row = mstart + m;
            bool ok = row < ce;
            int tok = ok ? mylist[row] : 0;
            float w = ok ? combine[tok * EEXP + e] : 0.0f;
            float val = (c[v] + b2[e * DDIM + d0 + nL]) * w;
            unsafeAtomicAdd(out + (size_t)tok * DDIM + d0 + nL, val);
        }
    }
}

// ---------------------------------------------------------------------------
// Kernel 3: aux load-balancing loss -> out[N*D]
// ---------------------------------------------------------------------------
__global__ void aux_kernel(const float* __restrict__ imp, const float* __restrict__ loadv,
                           float* __restrict__ out_aux) {
    float s = 0.0f;
#pragma unroll
    for (int e = 0; e < EEXP; ++e)
        s += (imp[e] / (float)N_TOK) * (loadv[e] / (float)N_TOK);
    out_aux[0] = (float)EEXP * s;
}

// ---------------------------------------------------------------------------
extern "C" void kernel_launch(void* const* d_in, const int* in_sizes, int n_in,
                              void* d_out, int out_size, void* d_ws, size_t ws_size,
                              hipStream_t stream) {
    const float* x  = (const float*)d_in[0];
    const float* gw = (const float*)d_in[1];
    const float* W1 = (const float*)d_in[2];
    const float* b1 = (const float*)d_in[3];
    const float* W2 = (const float*)d_in[4];
    const float* b2 = (const float*)d_in[5];
    float* out = (float*)d_out;

    // ws layout: [0..7]=cnt(int) [8..15]=imp [16..23]=load, combine[N*E] @ +32,
    // lists[E*N] after combine; packed weights at byte offset 256KB.
    int*   cnt     = (int*)d_ws;
    float* imp     = (float*)d_ws + 8;
    float* loadv   = (float*)d_ws + 16;
    float* combine = (float*)d_ws + 32;
    int*   lists   = (int*)((float*)d_ws + 32 + N_TOK * EEXP);

    const size_t W1P_BYTES = (size_t)EEXP * (FDIM / 16) * (DDIM / 32) * 1024; // 64 MB
    const size_t W2P_BYTES = (size_t)EEXP * (DDIM / 16) * (FDIM / 32) * 1024; // 64 MB
    const size_t PACK_OFF  = 256 * 1024;
    uint32_t* W1p = (uint32_t*)((char*)d_ws + PACK_OFF);
    uint32_t* W2p = (uint32_t*)((char*)d_ws + PACK_OFF + W1P_BYTES);
    const bool big_ws = ws_size >= PACK_OFF + W1P_BYTES + W2P_BYTES;

    zero_kernel<<<512, 256, 0, stream>>>(out, N_TOK * DDIM + 1, (float*)d_ws);
    gate_kernel<<<N_TOK / 8, 256, 0, stream>>>(x, gw, cnt, imp, loadv, combine, lists);

    if (big_ws) {
        pack_w1_kernel<<<8192, 256, 0, stream>>>(W1, W1p);
        pack_w2_kernel<<<8192, 256, 0, stream>>>(W2, W2p);
        dim3 grid2(N_TOK / MT2, EEXP);
        moe_packed_kernel<<<grid2, 256, 0, stream>>>(x, W1p, b1, W2p, b2,
                                                     cnt, combine, lists, out);
    } else {
        dim3 grid(N_TOK / MT, EEXP);
        moe_kernel<<<grid, 256, 0, stream>>>(x, W1, b1, W2, b2, cnt, combine, lists, out);
    }
    aux_kernel<<<1, 1, 0, stream>>>(imp, loadv, out + (size_t)N_TOK * DDIM);
}
